// feature_extractor_26688926777923
// MI455X (gfx1250) — matrix operand order repaired and checked by reading, not compiled, not run on hardware
//
#include <hip/hip_runtime.h>
#include <hip/hip_bf16.h>

typedef __attribute__((ext_vector_type(16))) _Float16 v16h;
typedef __attribute__((ext_vector_type(8)))  _Float16 v8h;
typedef __attribute__((ext_vector_type(8)))  float    v8f;
typedef unsigned int v4u __attribute__((ext_vector_type(4)));
typedef int          v4i __attribute__((ext_vector_type(4)));
typedef int          v8i __attribute__((ext_vector_type(8)));

#define NEG_INF (-3.4e38f)

// ---------------------------------------------------------------------------
// WMMA fragment helpers (CDNA5 16x16x32 f16 layouts, ISA 7.12.2)
// ---------------------------------------------------------------------------
__device__ __forceinline__ v16h make_frag(v8h lo, v8h hi) {
  v16h r;
#pragma unroll
  for (int i = 0; i < 8; ++i) { r[i] = lo[i]; r[i + 8] = hi[i]; }
  return r;
}

// A fragment: 16x32 (MxK) f16, row-major [M][ld]. lane&15 = M; lane>>4 picks K groups.
__device__ __forceinline__ v16h load_frag_a(const _Float16* base, int ld, int kk) {
  int lane = threadIdx.x & 31;
  int m = lane & 15, h = lane >> 4;
  const _Float16* p = base + m * ld + kk + 8 * h;
  v8h lo = *(const v8h*)p;         // K = 8h + 0..7
  v8h hi = *(const v8h*)(p + 16);  // K = 16 + 8h + 0..7
  return make_frag(lo, hi);
}

// B fragment: 32x16 (KxN) f16, activations stored [col][ld] (channel-contiguous).
__device__ __forceinline__ v16h load_frag_b(const _Float16* base, int ld, int kk) {
  int lane = threadIdx.x & 31;
  int n = lane & 15, h = lane >> 4;
  const _Float16* p = base + n * ld + kk + 8 * h;
  v8h lo = *(const v8h*)p;         // K = 8h + 0..7
  v8h hi = *(const v8h*)(p + 16);  // K = 16 + 8h + 0..7
  return make_frag(lo, hi);
}

// ---------------------------------------------------------------------------
// Kernel 1: weight conversion + BN folding
// sb layout (floats): s1[0:32] b1[32:64] s2[64:96] b2[96:128]
//                     s3[128:192] b3[192:256] s4[256:384] b4[384:512]
//                     s5[512:1024] b5[1024:1536]
// ---------------------------------------------------------------------------
__global__ void prep_kernel(const float* w1, const float* w2, const float* w3,
                            const float* w4, const float* w5,
                            const float* g1, const float* b1, const float* g2, const float* b2,
                            const float* g3, const float* b3, const float* g4, const float* b4,
                            const float* g5, const float* b5,
                            _Float16* w1h, _Float16* w2h, _Float16* w3h, _Float16* w4h,
                            _Float16* w5h, float* sb)
{
  const float RS = 0.99999500003749977f;  // 1/sqrt(1 + 1e-5)
  int g  = blockIdx.x * blockDim.x + threadIdx.x;
  int st = gridDim.x * blockDim.x;
  for (int i = g; i < 512 * 256; i += st) w5h[i] = (_Float16)w5[i];
  for (int i = g; i < 128 * 64;  i += st) w4h[i] = (_Float16)w4[i];
  for (int i = g; i < 64 * 32;   i += st) w3h[i] = (_Float16)w3[i];
  for (int i = g; i < 32 * 32;   i += st) {
    w2h[i] = (_Float16)w2[i];
    int o = i >> 5, kk = i & 31;                // w1 is [32,3] -> zero-pad K to 32
    w1h[i] = (kk < 3) ? (_Float16)w1[o * 3 + kk] : (_Float16)0.f;
  }
  for (int i = g; i < 32; i += st) {
    sb[i]      = g1[i] * RS;  sb[32 + i]  = b1[i];
    sb[64 + i] = g2[i] * RS;  sb[96 + i]  = b2[i];
  }
  for (int i = g; i < 64;  i += st) { sb[128 + i] = g3[i] * RS; sb[192 + i]  = b3[i]; }
  for (int i = g; i < 128; i += st) { sb[256 + i] = g4[i] * RS; sb[384 + i]  = b4[i]; }
  for (int i = g; i < 512; i += st) { sb[512 + i] = g5[i] * RS; sb[1024 + i] = b5[i]; }
}

// ---------------------------------------------------------------------------
// Kernel 2: KNN (top-32 of negative squared distance, self included, descending)
// One wave32 per point; 4096-entry distance row in LDS; 32 rounds of xor-shuffle
// argmax. Only the winning lane rescans its 128-entry stripe (incremental max).
// ---------------------------------------------------------------------------
__global__ __launch_bounds__(64) void knn_kernel(const float* __restrict__ x,
                                                 int* __restrict__ idx_flat,
                                                 float* __restrict__ knnp,
                                                 int* __restrict__ idxo)
{
  __shared__ float sdist[2][4096];
  int wave = threadIdx.x >> 5;
  int lane = threadIdx.x & 31;
  int p = blockIdx.x * 2 + wave;            // global point id, p < 32768
  int b = p >> 12, n = p & 4095;
  const float* xb = x + b * 3 * 4096;       // x layout [B][3][N]
  float cx = xb[n], cy = xb[4096 + n], cz = xb[8192 + n];
  float cs = cx * cx + cy * cy + cz * cz;
  float* dp = sdist[wave];

  float lmax = NEG_INF; int lidx = 0;
  for (int j = 0; j < 128; ++j) {
    int c = lane + (j << 5);
    float X = xb[c], Y = xb[4096 + c], Z = xb[8192 + c];
    float v = 2.f * (cx * X + cy * Y + cz * Z) - cs - (X * X + Y * Y + Z * Z);
    dp[c] = v;
    if (v > lmax) { lmax = v; lidx = c; }
  }

  int myK = 0;
  for (int r = 0; r < 32; ++r) {
    float v = lmax; int i = lidx;
#pragma unroll
    for (int off = 16; off; off >>= 1) {      // all lanes converge on argmax
      float ov = __shfl_xor(v, off, 32);
      int   oi = __shfl_xor(i, off, 32);
      if (ov > v || (ov == v && oi < i)) { v = ov; i = oi; }
    }
    if (lane == r) myK = i;
    if ((i & 31) == lane) {                   // only winner rescans its stripe
      dp[i] = NEG_INF;
      lmax = NEG_INF; lidx = 0;
      for (int j = 0; j < 128; ++j) {
        int c = lane + (j << 5);
        float dv = dp[c];
        if (dv > lmax) { lmax = dv; lidx = c; }
      }
    }
  }

  // lane k holds the k-th neighbor index (descending order, matches top_k)
  int gi = myK;
  float nx = xb[gi], ny = xb[4096 + gi], nz = xb[8192 + gi];
  int base = p * 32 + lane;
  idxo[base]     = gi;
  idx_flat[base] = gi + (b << 12);
  float* kp = knnp + base * 3;
  kp[0] = nx; kp[1] = ny; kp[2] = nz;
}

// ---------------------------------------------------------------------------
// Fused DGCNN stages 1-4. Max-over-k fused into the WMMA epilogue:
// every lane commits its 8 post-ReLU values with ds_max_num_f32 (the LDS
// atomic unit resolves the same-address conflicts; no bpermute/wait chains).
// ---------------------------------------------------------------------------
__device__ __forceinline__ void gemm_stage(int Cout, int CinP,
                                           const _Float16* inAct, _Float16* outAct,
                                           const _Float16* wts,
                                           const float* s, const float* bia,
                                           float* feat, int foff, bool storeAct)
{
  int wave = threadIdx.x >> 5;
  int lane = threadIdx.x & 31;
  int nl = lane & 15, h = lane >> 4;
  int ntiles = (Cout >> 4) * 4;                 // (Cout/16) x (64/16) output tiles
  for (int t = wave; t < ntiles; t += 4) {
    int mt = t >> 2, nt = t & 3;
    v8f acc = {};
    for (int kk = 0; kk < CinP; kk += 32) {
      v16h a  = load_frag_a(wts + mt * 16 * CinP, CinP, kk);
      v16h bf = load_frag_b(inAct + (nt * 16) * 128, 128, kk);
      acc = __builtin_amdgcn_wmma_f32_16x16x32_f16(false, a, false, bf,
                                                   (short)0, acc, false, false);
    }
    int col = nt * 16 + nl;
    int pt  = nt >> 1;                          // point within the 2-point workgroup
    float* fdst = feat + pt * 256 + foff;
#pragma unroll
    for (int r = 0; r < 8; ++r) {
      int m = mt * 16 + r + 8 * h;              // D layout: VGPR r -> M = r + 8*half
      float v = fmaxf(acc[r] * s[m] + bia[m], 0.f);
      if (storeAct) outAct[col * 128 + m] = (_Float16)v;
      atomicMax(&fdst[m], v);                   // ds_max_num_f32
    }
  }
}

__global__ __launch_bounds__(128) void dgcnn_fused_kernel(
    const float* __restrict__ x, const float* __restrict__ knnp,
    const _Float16* __restrict__ wAll,     // w1h|w2h|w3h|w4h contiguous, 12288 halves
    const float* __restrict__ sb, _Float16* __restrict__ feat256h)
{
  __shared__ __align__(16) _Float16 sW[12288];    // 24,576 B: w1|w2|w3|w4
  __shared__ __align__(16) _Float16 actA[64 * 128];
  __shared__ __align__(16) _Float16 actB[64 * 128];
  __shared__ float sSB[512];
  __shared__ __align__(16) float feat[2 * 256];

  int tid = threadIdx.x;
  int p0 = blockIdx.x * 2;

  // --- Tensor Data Mover: stage all 24.5 KB of weights into LDS (wave 0) ----
  if (tid < 32) {
    unsigned lds_off = (unsigned)(unsigned long long)(uintptr_t)(void*)sW;
    unsigned long long ga = (unsigned long long)(uintptr_t)wAll;
    v4u g0;
    g0[0] = 1u;                                       // count=1, user descriptor
    g0[1] = lds_off;                                  // D#.lds_addr
    g0[2] = (unsigned)(ga & 0xffffffffu);             // D#.global_addr[31:0]
    g0[3] = (unsigned)((ga >> 32) & 0x01ffffffu)      // D#.global_addr[56:32]
            | 0x80000000u;                            // type=2 ("image")
    v8i g1;
    g1[0] = 0x30000;                 // data_size=3 (8B), no pad/iterate/multicast
    g1[1] = (int)(3072u << 16);      // tensor_dim0[15:0]=3072 (8B units)
    g1[2] = 0x00010000;              // tensor_dim0[31:16]=0, tensor_dim1[15:0]=1
    g1[3] = (int)(3072u << 16);      // tensor_dim1[31:16]=0, tile_dim0=3072
    g1[4] = 1;                       // tile_dim1=1, tile_dim2=0
    g1[5] = 3072;                    // tensor_dim0_stride[31:0]
    g1[6] = 0;                       // stride hi / tensor_dim1_stride lo
    g1[7] = 0;
    v4i z4 = {0, 0, 0, 0};
    v8i z8 = {0, 0, 0, 0, 0, 0, 0, 0};
    __builtin_amdgcn_tensor_load_to_lds(g0, g1, z4, z4, z8, 0);
  }

  // --- overlap: BN params, zero-init, rel build while TDM streams weights ---
  for (int i = tid; i < 512;  i += 128) { sSB[i] = sb[i]; feat[i] = 0.f; }
  for (int i = tid; i < 8192; i += 128) actA[i] = (_Float16)0.f;
  __syncthreads();   // actA zeroed before rel fill touches it

  // Stage-1 input: rel = knn_point - center, channels 0..2 (3..31 stay zero)
  for (int i = tid; i < 192; i += 128) {
    int col = i / 3, c = i % 3;
    int p = p0 + (col >> 5);
    int b = p >> 12, n = p & 4095;
    float center = x[(b * 3 + c) * 4096 + n];
    float v = knnp[(p * 32 + (col & 31)) * 3 + c] - center;
    actA[col * 128 + c] = (_Float16)v;
  }
  if (tid < 32) __builtin_amdgcn_s_wait_tensorcnt(0);   // weights landed in LDS
  __syncthreads();

  const _Float16* sW1 = sW;
  const _Float16* sW2 = sW + 1024;
  const _Float16* sW3 = sW + 2048;
  const _Float16* sW4 = sW + 4096;

  gemm_stage(32,  32, actA, actB, sW1, sSB + 0,   sSB + 32,  feat, 0,   true);
  __syncthreads();
  gemm_stage(32,  32, actB, actA, sW2, sSB + 64,  sSB + 96,  feat, 32,  true);
  __syncthreads();
  gemm_stage(64,  32, actA, actB, sW3, sSB + 128, sSB + 192, feat, 64,  true);
  __syncthreads();
  gemm_stage(128, 64, actB, actA, sW4, sSB + 256, sSB + 384, feat, 128, false);
  __syncthreads();

  // vectorized store of the 2x256 concat vector (8 halves per thread)
  for (int i = tid; i < 64; i += 128) {
    v8h o;
#pragma unroll
    for (int e = 0; e < 8; ++e) o[e] = (_Float16)feat[i * 8 + e];
    *(v8h*)(feat256h + p0 * 256 + i * 8) = o;
  }
}

// ---------------------------------------------------------------------------
// Stage 5: features = relu(bn(w5 @ feat256)), M=512, K=256, N=32768.
// One wave per 16x64 strip: A fragment reused across 4 B tiles (32 WMMAs).
// ---------------------------------------------------------------------------
__global__ __launch_bounds__(128) void emb_kernel(const _Float16* __restrict__ feat256h,
                                                  const _Float16* __restrict__ w5h,
                                                  const float* __restrict__ sb,
                                                  float* __restrict__ features)
{
  int wid  = (blockIdx.x * 128 + threadIdx.x) >> 5;   // 16384 waves total
  int lane = threadIdx.x & 31;
  int mt = wid & 31;            // 512/16  = 32 row tiles
  int nb = wid >> 5;            // 32768/64 = 512 col blocks
  const float* s5 = sb + 512;
  const float* b5 = sb + 1024;

  v8f acc[4] = {{}, {}, {}, {}};
#pragma unroll
  for (int kk = 0; kk < 256; kk += 32) {
    v16h a = load_frag_a(w5h + mt * 16 * 256, 256, kk);
#pragma unroll
    for (int q = 0; q < 4; ++q) {
      v16h bf = load_frag_b(feat256h + (nb * 64 + q * 16) * 256, 256, kk);
      acc[q] = __builtin_amdgcn_wmma_f32_16x16x32_f16(false, a, false, bf,
                                                      (short)0, acc[q], false, false);
    }
  }

  int h = lane >> 4;
#pragma unroll
  for (int q = 0; q < 4; ++q) {
    int col = nb * 64 + q * 16 + (lane & 15);
    int b = col >> 12, n = col & 4095;
#pragma unroll
    for (int r = 0; r < 8; ++r) {
      int m = mt * 16 + r + 8 * h;
      float v = acc[q][r] * s5[m] + b5[m];
      v = v > 0.f ? v : 0.f;
      features[(b * 512 + m) * 4096 + n] = v;   // [B, EMB, N]
    }
  }
}

// ---------------------------------------------------------------------------
// Launch
// ---------------------------------------------------------------------------
extern "C" void kernel_launch(void* const* d_in, const int* in_sizes, int n_in,
                              void* d_out, int out_size, void* d_ws, size_t ws_size,
                              hipStream_t stream)
{
  (void)in_sizes; (void)n_in; (void)out_size; (void)ws_size;
  const float* x  = (const float*)d_in[0];
  const float* w1 = (const float*)d_in[1];
  const float* w2 = (const float*)d_in[2];
  const float* w3 = (const float*)d_in[3];
  const float* w4 = (const float*)d_in[4];
  const float* w5 = (const float*)d_in[5];
  const float* g1 = (const float*)d_in[6];  const float* b1 = (const float*)d_in[7];
  const float* g2 = (const float*)d_in[8];  const float* b2 = (const float*)d_in[9];
  const float* g3 = (const float*)d_in[10]; const float* b3 = (const float*)d_in[11];
  const float* g4 = (const float*)d_in[12]; const float* b4 = (const float*)d_in[13];
  const float* g5 = (const float*)d_in[14]; const float* b5 = (const float*)d_in[15];

  // workspace: feat256 (f16, 16 MB) + f16 weights (w1..w4 contiguous) + BN params
  char* ws = (char*)d_ws;
  _Float16* feat256h = (_Float16*)ws;                       // 32768*256 halves
  _Float16* w1h = (_Float16*)(ws + 16777216);               // start of 24,576 B block
  _Float16* w2h = w1h + 1024;
  _Float16* w3h = w2h + 1024;
  _Float16* w4h = w3h + 2048;
  _Float16* w5h = w4h + 8192;
  float*    sb  = (float*)(w5h + 131072);                   // 1536 floats

  // outputs: features | idx_flat | knn_points | idx (flat, in return order)
  float* features = (float*)d_out;                          // 8*512*4096
  int*   idx_flat = (int*)(features + 16777216);            // 8*4096*32
  float* knnp     = features + 16777216 + 1048576;          // 8*4096*32*3
  int*   idxo     = (int*)(features + 16777216 + 1048576 + 3145728);

  prep_kernel<<<128, 256, 0, stream>>>(w1, w2, w3, w4, w5,
                                       g1, b1, g2, b2, g3, b3, g4, b4, g5, b5,
                                       w1h, w2h, w3h, w4h, w5h, sb);
  knn_kernel<<<16384, 64, 0, stream>>>(x, idx_flat, knnp, idxo);
  dgcnn_fused_kernel<<<16384, 128, 0, stream>>>(x, knnp, w1h, sb, feat256h);
  emb_kernel<<<4096, 128, 0, stream>>>(feat256h, w5h, sb, features);
}
